// LinearAttentionCross_69277822485027
// MI455X (gfx1250) — matrix operand order repaired and checked by reading, not compiled, not run on hardware
//
#include <hip/hip_runtime.h>
#include <hip/hip_bf16.h>
#include <math.h>

// ---------------------------------------------------------------------------
// LinearAttentionCross on MI455X (gfx1250), bf16 WMMA pipeline.
// B=8, N=4096, DIM=C=512, HEADS=8, DIM_HEAD=64, HIDDEN=512.
// ---------------------------------------------------------------------------

typedef __attribute__((ext_vector_type(16))) __bf16 v16bf;
typedef __attribute__((ext_vector_type(8)))  float  v8f;

static __device__ __forceinline__ float b2f(unsigned short u) {
    unsigned int i = ((unsigned int)u) << 16;
    float f;
    __builtin_memcpy(&f, &i, 4);
    return f;
}
static __device__ __forceinline__ unsigned short f2b(float f) {
    unsigned int u;
    __builtin_memcpy(&u, &f, 4);
    unsigned int r = u + 0x7FFFu + ((u >> 16) & 1u);   // round-to-nearest-even
    return (unsigned short)(r >> 16);
}
// pack two fp32 -> one dword holding {bf16(lo), bf16(hi)}
static __device__ __forceinline__ unsigned int f2b_pk(float lo, float hi) {
    return (unsigned int)f2b(lo) | ((unsigned int)f2b(hi) << 16);
}

// Build a 16-element bf16 fragment from two 16-byte aligned runs.
static __device__ __forceinline__ v16bf load16(const unsigned short* p0,
                                               const unsigned short* p1) {
    uint4 a = *(const uint4*)p0;
    uint4 b = *(const uint4*)p1;
    v16bf r;
    __builtin_memcpy(&r, &a, 16);
    __builtin_memcpy(reinterpret_cast<char*>(&r) + 16, &b, 16);
    return r;
}

// ---------------------------------------------------------------------------
// Generic batched GEMM:  Out[b] = W[b] (OxC, bf16) * Act[b] (CxN) (+bias)
// Block tile: M=128, N=128, 256 threads = 8 waves.
// Wave w: m-tiles {2*(w&3), 2*(w&3)+1} x n-tiles {4*(w>>2) .. +3}  (8 acc).
// A fragment (16-bit A, 16x32): lane m=l&15, h=l>>4 holds K {8h..8h+7, 16+8h..+7}.
// B fragment: from LDS tile stored transposed [n][32]; lane n=l&15 holds
// K {8h..8h+7, 16+8h..+7} (same order as A).
// ---------------------------------------------------------------------------
__global__ __launch_bounds__(256) void gemm_wmma_bf16(
    const unsigned short* __restrict__ W, long long wBatchStride,
    const float* __restrict__ actF, const unsigned short* __restrict__ actB,
    unsigned short* __restrict__ outB, float* __restrict__ outF,
    const float* __restrict__ bias,
    int O, int C, int Nn)
{
    __shared__ unsigned short ldsA[128 * 32];   // [m][k]
    __shared__ unsigned short ldsB[128 * 32];   // [n][k]  (transposed act tile)

    const int b     = blockIdx.z;
    const int Mbase = blockIdx.y * 128;
    const int Nbase = blockIdx.x * 128;
    const int t     = threadIdx.x;
    const int w  = t >> 5, l  = t & 31;
    const int mt0 = (w & 3) * 2;                // first of two m-tiles
    const int ng  = w >> 2;                     // n-group (0/1)
    const int h2 = l >> 4, lm = l & 15;

    const unsigned short* Wb = W + (long long)b * wBatchStride;
    const size_t actOff = (size_t)b * C * Nn;
    const size_t outOff = (size_t)b * O * Nn;
    const bool useF = (actF != nullptr);

    v8f acc[2][4];
    #pragma unroll
    for (int mi = 0; mi < 2; ++mi)
        #pragma unroll
        for (int j = 0; j < 4; ++j)
            acc[mi][j] = (v8f){0.f,0.f,0.f,0.f,0.f,0.f,0.f,0.f};

    for (int k0 = 0; k0 < C; k0 += 32) {
        // ---- stage A tile [128 x 32] bf16: thread -> one row half (16 elems) ----
        {
            const int row = t >> 1, half = t & 1;
            const unsigned short* src = Wb + (size_t)(Mbase + row) * C + k0 + half * 16;
            uint4 a0 = *(const uint4*)src;
            uint4 a1 = *(const uint4*)(src + 8);
            *(uint4*)&ldsA[row * 32 + half * 16]     = a0;
            *(uint4*)&ldsA[row * 32 + half * 16 + 8] = a1;
        }
        // ---- stage B tile [32 x 128] transposed into LDS [128][32] ----
        // thread -> k-row pair (2r, 2r+1) x 8 columns; packed dword stores.
        {
            const int pr = t >> 4;              // k-row pair 0..15
            const int ch = t & 15;              // column octet 0..15
            const int n0 = Nbase + ch * 8;
            unsigned int* ldsB32 = (unsigned int*)ldsB;   // view [n][16] dwords
            if (useF) {
                const float* s0 = actF + actOff + (size_t)(k0 + 2 * pr) * Nn + n0;
                const float* s1 = s0 + Nn;
                if (k0 + 32 < C) __builtin_prefetch(s0 + 32 * Nn, 0, 3);
                #pragma unroll
                for (int i = 0; i < 8; ++i)
                    ldsB32[(n0 - Nbase + i) * 16 + pr] = f2b_pk(s0[i], s1[i]);
            } else {
                const unsigned short* s0 = actB + actOff + (size_t)(k0 + 2 * pr) * Nn + n0;
                const unsigned short* s1 = s0 + Nn;
                if (k0 + 32 < C) __builtin_prefetch(s0 + 32 * Nn, 0, 3);
                #pragma unroll
                for (int i = 0; i < 8; ++i)
                    ldsB32[(n0 - Nbase + i) * 16 + pr] =
                        (unsigned int)s0[i] | ((unsigned int)s1[i] << 16);
            }
        }
        __syncthreads();

        // ---- fragments + WMMA: 2 A frags, 4 B frags, 8 wmma ----
        v16bf af[2];
        #pragma unroll
        for (int mi = 0; mi < 2; ++mi) {
            const int am = ((mt0 + mi) * 16 + lm) * 32;
            af[mi] = load16(&ldsA[am + 8 * h2], &ldsA[am + 16 + 8 * h2]);
        }
        #pragma unroll
        for (int jj = 0; jj < 4; ++jj) {
            const int n = ((ng * 4 + jj) * 16 + lm) * 32;
            v16bf bf = load16(&ldsB[n + 8 * h2], &ldsB[n + 16 + 8 * h2]);
            #pragma unroll
            for (int mi = 0; mi < 2; ++mi)
                acc[mi][jj] = __builtin_amdgcn_wmma_f32_16x16x32_bf16(
                    false, af[mi], false, bf, (short)0, acc[mi][jj], false, false);
        }
        __syncthreads();
    }

    // ---- epilogue: C/D layout => VGPR r holds row (r + 8*h2), col = lm ----
    #pragma unroll
    for (int mi = 0; mi < 2; ++mi) {
        #pragma unroll
        for (int jj = 0; jj < 4; ++jj) {
            const int col = Nbase + (ng * 4 + jj) * 16 + lm;
            #pragma unroll
            for (int r = 0; r < 8; ++r) {
                const int row = Mbase + (mt0 + mi) * 16 + r + 8 * h2;
                const float v = acc[mi][jj][r];
                if (outF) outF[outOff + (size_t)row * Nn + col] = v + (bias ? bias[row] : 0.f);
                else      outB[outOff + (size_t)row * Nn + col] = f2b(v);
            }
        }
    }
}

// ---------------------------------------------------------------------------
// q softmax over head-dim (64 channels), times scale = 1/sqrt(64). In place, bf16.
// One thread per (b,h,n); loads coalesced across n.
// ---------------------------------------------------------------------------
__global__ __launch_bounds__(256) void softmax_q_kernel(unsigned short* __restrict__ q, int Nn)
{
    const long long t = (long long)blockIdx.x * 256 + threadIdx.x;   // b*8*Nn threads
    const int n  = (int)(t % Nn);
    const int bh = (int)(t / Nn);
    const int b = bh >> 3, h = bh & 7;
    unsigned short* base = q + ((size_t)(b * 512 + h * 64)) * Nn + n;

    float m = -1e30f;
    for (int d = 0; d < 64; ++d) m = fmaxf(m, b2f(base[(size_t)d * Nn]));
    float s = 0.f;
    for (int d = 0; d < 64; ++d) s += __expf(b2f(base[(size_t)d * Nn]) - m);
    const float inv = 0.125f / s;                       // scale = 64^-0.5 = 0.125
    for (int d = 0; d < 64; ++d) {
        const size_t idx = (size_t)d * Nn;
        base[idx] = f2b(__expf(b2f(base[idx]) - m) * inv);
    }
}

// ---------------------------------------------------------------------------
// k softmax over sequence n (4096), in place on rows [0..512) of KV per batch.
// One 256-thread block per row.
// ---------------------------------------------------------------------------
__global__ __launch_bounds__(256) void softmax_k_kernel(unsigned short* __restrict__ kv, int Nn)
{
    const int row = blockIdx.x;            // b*512 + c
    const int b = row >> 9, c = row & 511;
    unsigned short* p = kv + ((size_t)(b * 1024 + c)) * Nn;
    __shared__ float red[256];
    const int tid = threadIdx.x;

    float m = -1e30f;
    for (int i = tid; i < Nn; i += 256) m = fmaxf(m, b2f(p[i]));
    red[tid] = m; __syncthreads();
    for (int s = 128; s > 0; s >>= 1) {
        if (tid < s) red[tid] = fmaxf(red[tid], red[tid + s]);
        __syncthreads();
    }
    m = red[0]; __syncthreads();

    float sum = 0.f;
    for (int i = tid; i < Nn; i += 256) sum += __expf(b2f(p[i]) - m);
    red[tid] = sum; __syncthreads();
    for (int s = 128; s > 0; s >>= 1) {
        if (tid < s) red[tid] += red[tid + s];
        __syncthreads();
    }
    const float inv = 1.f / red[0];
    for (int i = tid; i < Nn; i += 256) p[i] = f2b(__expf(b2f(p[i]) - m) * inv);
}

// ---------------------------------------------------------------------------
// ctx[b,h,d,e] = sum_n k[b,h,d,n] * v[b,h,e,n].  Per (b,h): 64x64 out, K=4096.
// 16 waves; wave w -> tile (d-tile = w>>2, e-tile = w&3). Fragments loaded
// directly from global (rows of k / v are K-contiguous). f32 output.
// ---------------------------------------------------------------------------
__global__ __launch_bounds__(512) void ctx_wmma_kernel(
    const unsigned short* __restrict__ kv, float* __restrict__ ctx, int Nn)
{
    const int bh = blockIdx.x;
    const int b = bh >> 3, h = bh & 7;
    const int t = threadIdx.x;
    const int w = t >> 5, l = t & 31;
    const int ti = w >> 2, tj = w & 3;
    const int h2 = l >> 4, lm = l & 15;

    const unsigned short* arow =
        kv + ((size_t)(b * 1024 +       h * 64 + ti * 16 + lm)) * Nn;   // k row d
    const unsigned short* brow =
        kv + ((size_t)(b * 1024 + 512 + h * 64 + tj * 16 + lm)) * Nn;   // v row e

    v8f acc = (v8f){0.f,0.f,0.f,0.f,0.f,0.f,0.f,0.f};
    for (int k0 = 0; k0 < Nn; k0 += 32) {
        v16bf af = load16(arow + k0 + 8 * h2,  arow + k0 + 16 + 8 * h2);
        v16bf bf = load16(brow + k0 + 8 * h2,  brow + k0 + 16 + 8 * h2);
        acc = __builtin_amdgcn_wmma_f32_16x16x32_bf16(
            false, af, false, bf, (short)0, acc, false, false);
    }
    #pragma unroll
    for (int r = 0; r < 8; ++r) {
        const int d = ti * 16 + r + 8 * h2;
        const int e = tj * 16 + lm;
        ctx[((size_t)(bh * 64 + d)) * 64 + e] = acc[r];
    }
}

// ---------------------------------------------------------------------------
// M_big[b, o, h*64+d] = sum_e Wout[o, h*64+e] * ctx[b,h,d,e]   (bf16 out)
// Folds Wout into the per-head ctx so the final stage is one 512x512 GEMM.
// ---------------------------------------------------------------------------
__global__ __launch_bounds__(256) void mbig_kernel(
    const float* __restrict__ Wout, const float* __restrict__ ctx,
    unsigned short* __restrict__ M)
{
    const int t = blockIdx.x * 256 + threadIdx.x;    // 8*512*512 threads
    const int c2 = t & 511;
    const int o  = (t >> 9) & 511;
    const int b  = t >> 18;
    const int h = c2 >> 6, d = c2 & 63;
    const float* wrow = Wout + (size_t)o * 512 + h * 64;
    const float* crow = ctx + ((size_t)((b * 8 + h) * 64 + d)) * 64;
    float s = 0.f;
    #pragma unroll 8
    for (int e = 0; e < 64; ++e) s += wrow[e] * crow[e];
    M[((size_t)(b * 512) + o) * 512 + c2] = f2b(s);
}

// ---------------------------------------------------------------------------
// In-place LayerNorm over channels (512) per (b,n), then * g[c].
// ---------------------------------------------------------------------------
__global__ __launch_bounds__(256) void layernorm_kernel(
    float* __restrict__ y, const float* __restrict__ g, int Nn)
{
    const int t = blockIdx.x * 256 + threadIdx.x;   // b*Nn threads
    const int n = t % Nn, b = t / Nn;
    float* base = y + (size_t)b * 512 * Nn + n;
    float s = 0.f, ss = 0.f;
    for (int c = 0; c < 512; ++c) {
        const float v = base[(size_t)c * Nn];
        s += v; ss += v * v;
    }
    const float mean = s * (1.f / 512.f);
    const float var  = ss * (1.f / 512.f) - mean * mean;
    const float inv  = rsqrtf(var + 1e-5f);
    for (int c = 0; c < 512; ++c) {
        const size_t idx = (size_t)c * Nn;
        base[idx] = (base[idx] - mean) * inv * g[c];
    }
}

// ---------------------------------------------------------------------------
// fp32 -> bf16 weight conversion
// ---------------------------------------------------------------------------
__global__ __launch_bounds__(256) void cvt_f32_bf16(
    const float* __restrict__ src, unsigned short* __restrict__ dst, int n)
{
    const int i = blockIdx.x * 256 + threadIdx.x;
    if (i < n) dst[i] = f2b(src[i]);
}

// ---------------------------------------------------------------------------
extern "C" void kernel_launch(void* const* d_in, const int* in_sizes, int n_in,
                              void* d_out, int out_size, void* d_ws, size_t ws_size,
                              hipStream_t stream)
{
    const float* x    = (const float*)d_in[0];
    const float* ctxI = (const float*)d_in[1];
    const float* Wq   = (const float*)d_in[2];
    const float* Wkv  = (const float*)d_in[3];
    const float* Wout = (const float*)d_in[4];
    const float* bout = (const float*)d_in[5];
    const float* g    = (const float*)d_in[6];
    float* out = (float*)d_out;

    const int Bn = 8, Nn = 4096, C = 512;

    // workspace layout (~108 MB)
    char* ws = (char*)d_ws;
    unsigned short* wqB  = (unsigned short*)ws;                         // 512*512
    unsigned short* wkvB = wqB + (size_t)512 * 512;                     // 1024*512
    unsigned short* qB   = wkvB + (size_t)1024 * 512;                   // 8*512*4096
    unsigned short* kvB  = qB + (size_t)Bn * 512 * Nn;                  // 8*1024*4096
    float* ctxW          = (float*)(kvB + (size_t)Bn * 1024 * Nn);      // 8*8*64*64
    unsigned short* mB   = (unsigned short*)(ctxW + (size_t)Bn * 8 * 64 * 64); // 8*512*512

    // 1) weight conversion
    cvt_f32_bf16<<<(512 * 512 + 255) / 256, 256, 0, stream>>>(Wq, wqB, 512 * 512);
    cvt_f32_bf16<<<(1024 * 512 + 255) / 256, 256, 0, stream>>>(Wkv, wkvB, 1024 * 512);

    // 2) Q = Wq*x, KV = Wkv*context  (bf16 out)
    dim3 g1(Nn / 128, 512 / 128, Bn);
    gemm_wmma_bf16<<<g1, 256, 0, stream>>>(wqB, 0, x, nullptr, qB, nullptr, nullptr,
                                           512, C, Nn);
    dim3 g2(Nn / 128, 1024 / 128, Bn);
    gemm_wmma_bf16<<<g2, 256, 0, stream>>>(wkvB, 0, ctxI, nullptr, kvB, nullptr, nullptr,
                                           1024, C, Nn);

    // 3) softmaxes (in place)
    softmax_q_kernel<<<(Bn * 8 * Nn) / 256, 256, 0, stream>>>(qB, Nn);
    softmax_k_kernel<<<Bn * 512, 256, 0, stream>>>(kvB, Nn);

    // 4) ctx = k * v^T per (b,h)
    ctx_wmma_kernel<<<Bn * 8, 512, 0, stream>>>(kvB, ctxW, Nn);

    // 5) fold Wout into ctx: M_big = Wout * ctx^T  (per batch, bf16)
    mbig_kernel<<<(Bn * 512 * 512) / 256, 256, 0, stream>>>(Wout, ctxW, mB);

    // 6) final = M_big * q_sm + b_out  (fp32 straight into d_out)
    gemm_wmma_bf16<<<g1, 256, 0, stream>>>(mB, (long long)512 * 512, nullptr, qB,
                                           nullptr, out, bout, 512, C, Nn);

    // 7) LayerNorm in place on d_out
    layernorm_kernel<<<(Bn * Nn) / 256, 256, 0, stream>>>(out, g, Nn);
}
